// sGCN_33311766348266
// MI455X (gfx1250) — compile-verified, matrix operand order checked by reading, not run on hardware
//
#include <hip/hip_runtime.h>
#include <stdint.h>

#define NN    1024
#define LDT   40          // LDS row stride in bf16 elements (32 + 8 pad)
#define TILE  (128 * LDT) // elements per LDS tile buffer

typedef __attribute__((ext_vector_type(16))) __bf16 v16bf;
typedef __attribute__((ext_vector_type(8)))  float  v8f;
typedef __attribute__((ext_vector_type(2)))  int    v2i;

#if __has_builtin(__builtin_amdgcn_global_load_async_to_lds_b64) && \
    __has_builtin(__builtin_amdgcn_s_wait_asynccnt)
#define USE_ASYNC_LDS 1
typedef __attribute__((address_space(1))) v2i* gv2i_ptr;
typedef __attribute__((address_space(3))) v2i* lv2i_ptr;
#else
#define USE_ASYNC_LDS 0
#endif

__device__ __forceinline__ unsigned short f2bf(float f) {
  unsigned int u = __float_as_uint(f);
  u += 0x7FFFu + ((u >> 16) & 1u);
  return (unsigned short)(u >> 16);
}
__device__ __forceinline__ float bf2f(unsigned short h) {
  return __uint_as_float(((unsigned int)h) << 16);
}

// Pack two f32 -> bf16 pair. Preferred: cvt_pk_bf16_f32 builtin; else 2x v_add + 1x v_perm_b32.
__device__ __forceinline__ unsigned int pk_bf16(float lo, float hi) {
#if __has_builtin(__builtin_amdgcn_cvt_pk_bf16_f32)
  typedef __attribute__((ext_vector_type(2))) __bf16 v2bf;
  union { v2bf v; unsigned int u; } c;
  c.v = __builtin_amdgcn_cvt_pk_bf16_f32(lo, hi);
  return c.u;
#elif __has_builtin(__builtin_amdgcn_perm)
  // round-half-up then take high 16 bits of each; merge with one v_perm_b32
  return __builtin_amdgcn_perm(__float_as_uint(hi) + 0x8000u,
                               __float_as_uint(lo) + 0x8000u, 0x07060302u);
#else
  return (unsigned int)f2bf(lo) | ((unsigned int)f2bf(hi) << 16);
#endif
}

union Frag16 { uint4 q[2]; v16bf v; };

// bf16 fragment from an LDS row (K-contiguous). Per ISA layout:
// lanes 0-15 hold K = kbase+0..7 (regs 0-3) and kbase+16..23 (regs 4-7),
// lanes 16-31 use kbase += 8. Two 16B ds loads.
__device__ __forceinline__ v16bf load_frag_lds(const unsigned short* rowbase, int kbase) {
  Frag16 f;
  f.q[0] = *reinterpret_cast<const uint4*>(rowbase + kbase);
  f.q[1] = *reinterpret_cast<const uint4*>(rowbase + kbase + 16);
  return f.v;
}

// Load a 128x32 f32 tile into 4 float4 regs per thread (optional K guard, K%4==0).
template <bool GUARD>
__device__ __forceinline__ void load_tile_f32(const float* __restrict__ tile0,
                                              int srcStride, int tid, int k0, int K,
                                              bool pf, float4 (&r)[4]) {
#pragma unroll
  for (int it = 0; it < 4; ++it) {
    const int slot = it * 256 + tid;
    const int row = slot >> 3, kq = (slot & 7) * 4;
    const float* src = tile0 + (size_t)row * srcStride + k0 + kq;
    r[it] = make_float4(0.f, 0.f, 0.f, 0.f);
    if (!GUARD || (k0 + kq < K)) {
      r[it] = *reinterpret_cast<const float4*>(src);
      if (pf) __builtin_prefetch(src + 32, 0, 1);
    }
  }
}

__device__ __forceinline__ void store_tile_bf16(const float4 (&r)[4],
                                                unsigned short* lds, int tid) {
#pragma unroll
  for (int it = 0; it < 4; ++it) {
    const int slot = it * 256 + tid;
    const int row = slot >> 3, kq = (slot & 7) * 4;
    *reinterpret_cast<uint2*>(&lds[row * LDT + kq]) =
        make_uint2(pk_bf16(r[it].x, r[it].y), pk_bf16(r[it].z, r[it].w));
  }
}

// One K-step: 2 A frags + 4 B frags -> 8 WMMAs (wave tile 32x64).
__device__ __forceinline__ void compute_tiles(const unsigned short* As,
                                              const unsigned short* Bs,
                                              int wm, int wn, int lr, int hi,
                                              v8f (&acc)[2][4]) {
  v16bf aF[2], bF[4];
#pragma unroll
  for (int tm = 0; tm < 2; ++tm)
    aF[tm] = load_frag_lds(&As[(wm + tm * 16 + lr) * LDT], hi * 8);
#pragma unroll
  for (int tn = 0; tn < 4; ++tn)
    bF[tn] = load_frag_lds(&Bs[(wn + tn * 16 + lr) * LDT], hi * 8);
#pragma unroll
  for (int tm = 0; tm < 2; ++tm)
#pragma unroll
    for (int tn = 0; tn < 4; ++tn)
      acc[tm][tn] = __builtin_amdgcn_wmma_f32_16x16x32_bf16(
          false, aF[tm], false, bF[tn], (short)0, acc[tm][tn], false, false);
}

// ---------------------------------------------------------------------------
// Kernel 1: dinv[b][j] = rsqrt(sum_i Ghat[b][i][j]),  inf -> 0
// ---------------------------------------------------------------------------
__global__ void degree_kernel(const float* __restrict__ G,
                              const int* __restrict__ seq,
                              float* __restrict__ dinv) {
  const int j = blockIdx.x * blockDim.x + threadIdx.x;
  const int b = blockIdx.y;
  const float* Gb = G + (size_t)b * NN * NN;
  float s = 0.f;
  for (int i = 0; i < NN; ++i) s += Gb[(size_t)i * NN + j];
  if (j < seq[b]) s += 1.0f;
  dinv[b * NN + j] = (s > 0.f) ? rsqrtf(s) : 0.f;
}

// ---------------------------------------------------------------------------
// Kernel 2/4: YT[b][f][j] = bf16( dinv[b][j] * (X[b] @ W^T)[j][f] )
//   Double-buffered LDS, one barrier per K-tile; wave tile 32x64.
// ---------------------------------------------------------------------------
__global__ void __launch_bounds__(256, 1)
gemm_feat_kernel(const float* __restrict__ X,
                 const float* __restrict__ W,
                 const float* __restrict__ dinv,
                 unsigned short* __restrict__ YT,
                 int K, int F) {
  __shared__ unsigned short As[2 * TILE];
  __shared__ unsigned short Bs[2 * TILE];
  const int tid  = threadIdx.x;
  const int b    = blockIdx.z;
  const int mBlk = blockIdx.x * 128;
  const int nBlk = blockIdx.y * 128;
  const int wave = tid >> 5, lane = tid & 31;
  const int lr = lane & 15, hi = lane >> 4;
  const int wm = (wave & 3) * 32;
  const int wn = (wave >> 2) * 64;
  const float* Xt = X + (size_t)b * NN * K + (size_t)mBlk * K;
  const float* Wt = W + (size_t)nBlk * K;

  v8f acc[2][4];
#pragma unroll
  for (int i = 0; i < 2; ++i)
#pragma unroll
    for (int j = 0; j < 4; ++j)
#pragma unroll
      for (int e = 0; e < 8; ++e) acc[i][j][e] = 0.f;

  const int T = (K + 31) >> 5;         // total K tiles (last may be partial)
  float4 ar[4], br[4];
  load_tile_f32<false>(Xt, K, tid, 0, K, true, ar);   // t=0 is always full
  load_tile_f32<false>(Wt, K, tid, 0, K, false, br);

  int cur = 0;
  for (int t = 0; t < T; ++t) {
    store_tile_bf16(ar, As + cur * TILE, tid);
    store_tile_bf16(br, Bs + cur * TILE, tid);
    __syncthreads();
    if (t + 1 < T) {
      const int nk0 = (t + 1) * 32;
      if (nk0 + 32 <= K) {
        load_tile_f32<false>(Xt, K, tid, nk0, K, t + 2 < T, ar);
        load_tile_f32<false>(Wt, K, tid, nk0, K, false, br);
      } else {
        load_tile_f32<true>(Xt, K, tid, nk0, K, false, ar);
        load_tile_f32<true>(Wt, K, tid, nk0, K, false, br);
      }
    }
    compute_tiles(As + cur * TILE, Bs + cur * TILE, wm, wn, lr, hi, acc);
    cur ^= 1;
  }

  // Epilogue: scale row j by dinv[j], store transposed, 16B per lane per tile.
  const float* db = dinv + b * NN;
#pragma unroll
  for (int tm = 0; tm < 2; ++tm) {
    const int j0 = mBlk + wm + tm * 16 + hi * 8;
#pragma unroll
    for (int tn = 0; tn < 4; ++tn) {
      const int f = nBlk + wn + tn * 16 + lr;
      unsigned int u[4];
#pragma unroll
      for (int q = 0; q < 4; ++q)
        u[q] = pk_bf16(db[j0 + 2 * q]     * acc[tm][tn][2 * q],
                       db[j0 + 2 * q + 1] * acc[tm][tn][2 * q + 1]);
      uint4 st; st.x = u[0]; st.y = u[1]; st.z = u[2]; st.w = u[3];
      *reinterpret_cast<uint4*>(&YT[((size_t)b * F + f) * NN + j0]) = st;
    }
  }
}

// ---------------------------------------------------------------------------
// Kernel 3/5: Out[b][i][f] = relu( dinv[i]*( sum_j G[i][j]*Ys[j][f]
//                                          + mask[i]*Ys[i][f] ) + bias[f] )
//   Double-buffered; B tile via async global->LDS (ASYNCcnt) when available.
// ---------------------------------------------------------------------------
__global__ void __launch_bounds__(256, 1)
gemm_graph_kernel(const float* __restrict__ G,
                  const unsigned short* __restrict__ YT,
                  const float* __restrict__ dinv,
                  const float* __restrict__ bias,
                  const int* __restrict__ seq,
                  float* __restrict__ Out,
                  int F) {
  __shared__ unsigned short As[2 * TILE];
  __shared__ unsigned short Bs[2 * TILE];
  const int tid  = threadIdx.x;
  const int b    = blockIdx.z;
  const int mBlk = blockIdx.x * 128;
  const int nBlk = blockIdx.y * 128;
  const int wave = tid >> 5, lane = tid & 31;
  const int lr = lane & 15, hi = lane >> 4;
  const int wm = (wave & 3) * 32;
  const int wn = (wave >> 2) * 64;
  const float* Gt = G + (size_t)b * NN * NN + (size_t)mBlk * NN;
  const unsigned short* Yb = YT + (size_t)b * F * NN;
  const unsigned short* Yt = Yb + (size_t)nBlk * NN;

  v8f acc[2][4];
#pragma unroll
  for (int i = 0; i < 2; ++i)
#pragma unroll
    for (int j = 0; j < 4; ++j)
#pragma unroll
      for (int e = 0; e < 8; ++e) acc[i][j][e] = 0.f;

  const int T = NN / 32;
  float4 ar[4];

  // B-tile staging: async writes straight to LDS, else stage through regs.
#if USE_ASYNC_LDS
#define ISSUE_B(k0, buf)                                                       \
  {                                                                            \
    _Pragma("unroll") for (int it = 0; it < 4; ++it) {                         \
      const int slot = it * 256 + tid;                                         \
      const int n = slot >> 3, kq = (slot & 7) * 4;                            \
      __builtin_amdgcn_global_load_async_to_lds_b64(                           \
          (gv2i_ptr)(const_cast<unsigned short*>(Yt + (size_t)n * NN + (k0) + kq)), \
          (lv2i_ptr)&(buf)[n * LDT + kq], 0, 0);                               \
    }                                                                          \
  }
  ISSUE_B(0, Bs);
#else
  uint2 brr[4];
#define LOAD_B(k0)                                                             \
  {                                                                            \
    _Pragma("unroll") for (int it = 0; it < 4; ++it) {                         \
      const int slot = it * 256 + tid;                                         \
      const int n = slot >> 3, kq = (slot & 7) * 4;                            \
      brr[it] = *reinterpret_cast<const uint2*>(Yt + (size_t)n * NN + (k0) + kq); \
    }                                                                          \
  }
#define STORE_B(buf)                                                           \
  {                                                                            \
    _Pragma("unroll") for (int it = 0; it < 4; ++it) {                         \
      const int slot = it * 256 + tid;                                         \
      const int n = slot >> 3, kq = (slot & 7) * 4;                            \
      *reinterpret_cast<uint2*>(&(buf)[n * LDT + kq]) = brr[it];               \
    }                                                                          \
  }
  LOAD_B(0);
#endif
  load_tile_f32<false>(Gt, NN, tid, 0, NN, true, ar);

  int cur = 0;
  for (int t = 0; t < T; ++t) {
    store_tile_bf16(ar, As + cur * TILE, tid);
#if USE_ASYNC_LDS
    __builtin_amdgcn_s_wait_asynccnt(0);   // B tile t resident in LDS
#else
    STORE_B(Bs + cur * TILE);
#endif
    __syncthreads();
    if (t + 1 < T) {
      const int nk0 = (t + 1) * 32;
#if USE_ASYNC_LDS
      ISSUE_B(nk0, Bs + (cur ^ 1) * TILE);
#else
      LOAD_B(nk0);
#endif
      load_tile_f32<false>(Gt, NN, tid, nk0, NN, t + 2 < T, ar);
    }
    compute_tiles(As + cur * TILE, Bs + cur * TILE, wm, wn, lr, hi, acc);
    cur ^= 1;
  }

  // Epilogue: diagonal correction + row scale + bias + relu.
  const int slen = seq[b];
  const float* db = dinv + b * NN;
#pragma unroll
  for (int tm = 0; tm < 2; ++tm) {
#pragma unroll
    for (int tn = 0; tn < 4; ++tn) {
      const int f = nBlk + wn + tn * 16 + lr;
#pragma unroll
      for (int r = 0; r < 8; ++r) {
        const int i = mBlk + wm + tm * 16 + hi * 8 + r;
        float v = acc[tm][tn][r];
        if (i < slen) v += bf2f(Yb[(size_t)f * NN + i]);  // mask[i]*Ys[i][f]
        const float o = db[i] * v + bias[f];
        Out[((size_t)b * NN + i) * F + f] = fmaxf(o, 0.f);
      }
    }
  }
}

// ---------------------------------------------------------------------------
extern "C" void kernel_launch(void* const* d_in, const int* in_sizes, int n_in,
                              void* d_out, int out_size, void* d_ws, size_t ws_size,
                              hipStream_t stream) {
  (void)in_sizes; (void)n_in; (void)out_size; (void)ws_size;
  const float* x  = (const float*)d_in[0];   // [32][1024][300]
  const float* G  = (const float*)d_in[1];   // [32][1024][1024]
  const int*  seq = (const int*)  d_in[2];   // [32]
  const float* W1 = (const float*)d_in[3];   // [256][300]
  const float* b1 = (const float*)d_in[4];   // [256]
  const float* W2 = (const float*)d_in[5];   // [128][256]
  const float* b2 = (const float*)d_in[6];   // [128]
  float* out = (float*)d_out;                // [32][1024][128]

  char* ws = (char*)d_ws;
  float*          dinv = (float*)ws;                                  // 128 KB
  unsigned short* YT1  = (unsigned short*)(ws + 131072);              // 16 MB
  float*          H1   = (float*)(ws + 131072 + 16777216);            // 32 MB
  unsigned short* YT2  = (unsigned short*)(ws + 131072 + 16777216 + 33554432); // 8 MB

  degree_kernel<<<dim3(NN / 256, 32), 256, 0, stream>>>(G, seq, dinv);
  gemm_feat_kernel<<<dim3(8, 2, 32), 256, 0, stream>>>(x, W1, dinv, YT1, 300, 256);
  gemm_graph_kernel<<<dim3(8, 2, 32), 256, 0, stream>>>(G, YT1, dinv, b1, seq, H1, 256);
  gemm_feat_kernel<<<dim3(8, 1, 32), 256, 0, stream>>>(H1, W2, dinv, YT2, 256, 128);
  gemm_graph_kernel<<<dim3(8, 1, 32), 256, 0, stream>>>(G, YT2, dinv, b2, seq, out, 128);
}
